// MentionProposalModule_73933567033622
// MI455X (gfx1250) — matrix operand order repaired and checked by reading, not compiled, not run on hardware
//
#include <hip/hip_runtime.h>
#include <stdint.h>

// ---------------- problem constants ----------------
#define NW    4096
#define H     768
#define NC    49152
#define MSW   30
#define E     20
#define MH    1024            // mlp hidden
#define KSPAN 2324            // 3*H + E
#define KTILES 73             // ceil(KSPAN/32)
#define KPAD  (KTILES*32)     // 2336
#define NTILES 64             // MH/16
#define ROWS  64              // candidates per block (four 16-row WMMA tiles)
#define WAVES 16              // 512 threads
#define RT    4               // row tiles per block
#define NTW   4               // n-tiles per wave (16 waves * 4 = 64)
#define ROW_DW 1172           // padded A-row stride in dwords (2344 bf16), 16B aligned,
                              // (1172 % 64)=20 -> conflict-free ds_load_b128 across 16 rows

typedef __bf16   bf16x16 __attribute__((ext_vector_type(16)));
typedef float    f32x8   __attribute__((ext_vector_type(8)));
typedef uint32_t u32x4   __attribute__((ext_vector_type(4)));

struct Frag { u32x4 lo; u32x4 hi; };   // 32 bytes == one 16x32 / 32x16 bf16 fragment per lane

static __device__ __forceinline__ uint16_t f2bf(float f) {
  uint32_t u = __builtin_bit_cast(uint32_t, f);
  uint32_t r = (u + 0x7FFFu + ((u >> 16) & 1u)) >> 16;   // round-to-nearest-even
  return (uint16_t)r;
}

// ---------------- kernel 1: word_attn[NW] = doc @ attn_w + attn_b ----------------
__global__ void k_word_attn(const float* __restrict__ doc, const float* __restrict__ w,
                            const float* __restrict__ b, float* __restrict__ out) {
  int wave = threadIdx.x >> 5, lane = threadIdx.x & 31;
  int word = blockIdx.x * 8 + wave;
  const float* row = doc + word * H;
  float s = 0.f;
  #pragma unroll
  for (int j = 0; j < H / 32; ++j) s += row[lane + 32 * j] * w[lane + 32 * j];
  #pragma unroll
  for (int m = 16; m; m >>= 1) s += __shfl_xor(s, m, 32);
  if (lane == 0) out[word] = s + b[0];
}

// ---------------- kernel 2: 30-entry width score table ----------------
__global__ void k_width_scores(const float* __restrict__ wp_emb, const float* __restrict__ w1,
                               const float* __restrict__ b1, const float* __restrict__ w2,
                               const float* __restrict__ b2, float* __restrict__ out) {
  __shared__ float red[256];
  int w = blockIdx.x;
  float wp[E];
  #pragma unroll
  for (int k = 0; k < E; ++k) wp[k] = wp_emb[w * E + k];
  float acc = 0.f;
  for (int j = threadIdx.x; j < MH; j += 256) {
    float h = b1[j];
    #pragma unroll
    for (int k = 0; k < E; ++k) h += wp[k] * w1[k * MH + j];
    acc += fmaxf(h, 0.f) * w2[j];
  }
  red[threadIdx.x] = acc;
  __syncthreads();
  for (int s = 128; s; s >>= 1) {
    if (threadIdx.x < (unsigned)s) red[threadIdx.x] += red[threadIdx.x + s];
    __syncthreads();
  }
  if (threadIdx.x == 0) out[w] = red[0] + b2[0];
}

// ---------------- kernel 3: pack mlp1_w (fp32 KxN) into bf16 B-fragment layout ----------------
// Tile (ktile, ntile) = 32x16 bf16. Contiguous k order, same as the A reads: lanes 0-15 hold K=0..15 (2/VGPR),
// lanes 16-31 hold K=16..31, column n = lane&15. Stored so each lane reads 8 contiguous dwords.
__global__ void k_pack_w1(const float* __restrict__ w1, uint32_t* __restrict__ dst) {
  int tile  = blockIdx.x;            // ktile*64 + ntile
  int ktile = tile >> 6, ntile = tile & 63;
  int lane  = threadIdx.x >> 3, v = threadIdx.x & 7;
  int n = ntile * 16 + (lane & 15);
  int k = ktile * 32 + (lane >> 4) * 16 + 2 * v;
  uint32_t d = 0;
  if (k < KSPAN) {  // KSPAN even -> pair never straddles the boundary
    uint32_t lo = f2bf(w1[k * MH + n]);
    uint32_t hi = f2bf(w1[(k + 1) * MH + n]);
    d = lo | (hi << 16);
  }
  dst[tile * 256 + threadIdx.x] = d;
}

// ---------------- kernel 4: fused span-emb build + WMMA GEMM + scoring ----------------
// 512 threads = 16 waves. Block computes 64 candidates; each wave owns a 64-column strip
// and all four 16-row tiles, so every B fragment feeds 4 WMMAs (B L2 traffic halved vs 2).
__global__ __launch_bounds__(512, 1)
void k_main(const float* __restrict__ doc, const int* __restrict__ starts,
            const int* __restrict__ ends, const float* __restrict__ width_emb,
            const float* __restrict__ word_attn, const uint32_t* __restrict__ Bp,
            const float* __restrict__ mlp1_b, const float* __restrict__ mlp2_w,
            const float* __restrict__ mlp2_b, const float* __restrict__ wscore,
            float* __restrict__ out) {
  extern __shared__ char smem[];
  uint16_t* A       = (uint16_t*)smem;                                 // ROWS x ROW_DW dwords (bf16)
  float*    partial = (float*)(smem + (size_t)ROWS * ROW_DW * 4);      // [WAVES][ROWS]
  int*      widxs   = (int*)(smem + (size_t)ROWS * ROW_DW * 4 + (size_t)WAVES * ROWS * 4);

  const int wave = threadIdx.x >> 5;
  const int lane = threadIdx.x & 31;
  const int m0   = lane & 15;
  const int half = lane >> 4;

  // ---- phase 1: each wave builds 4 span-embedding rows in LDS (bf16) ----
  for (int rr = 0; rr < 4; ++rr) {
    int r = wave * 4 + rr;
    int c = blockIdx.x * ROWS + r;
    int s = starts[c], e = ends[c];
    int wd = e - s; wd = wd < 0 ? 0 : (wd > MSW - 1 ? MSW - 1 : wd);
    int len = e - s + 1; if (len < 1) len = 1; if (len > 32) len = 32;
    uint16_t* arow = A + (size_t)r * (ROW_DW * 2);

    // masked softmax over span (span length <= 30 < 32 lanes)
    float a = (lane < len) ? word_attn[s + lane] : -3.0e38f;
    float mx = a;
    #pragma unroll
    for (int m = 16; m; m >>= 1) mx = fmaxf(mx, __shfl_xor(mx, m, 32));
    float p = (lane < len) ? __expf(a - mx) : 0.f;
    float sm = p;
    #pragma unroll
    for (int m = 16; m; m >>= 1) sm += __shfl_xor(sm, m, 32);
    float pn = p / sm;

    // attention term: each lane owns 24 columns
    float at[H / 32];
    #pragma unroll
    for (int j = 0; j < H / 32; ++j) at[j] = 0.f;
    for (int wi = 0; wi < len; ++wi) {
      float pw = __shfl(pn, wi, 32);
      const float* drow = doc + (size_t)(s + wi) * H;
      #pragma unroll
      for (int j = 0; j < H / 32; ++j) at[j] += pw * drow[lane + 32 * j];
    }

    const float* srow = doc + (size_t)s * H;
    const float* erow = doc + (size_t)e * H;
    #pragma unroll
    for (int j = 0; j < H / 32; ++j) {
      int col = lane + 32 * j;
      arow[col]             = f2bf(srow[col]);      // start_emb  [0,768)
      arow[H + col]         = f2bf(erow[col]);      // end_emb    [768,1536)
      arow[2 * H + E + col] = f2bf(at[j]);          // attn term  [1556,2324)
    }
    if (lane < E)            arow[2 * H + lane] = f2bf(width_emb[wd * E + lane]); // [1536,1556)
    if (lane < KPAD - KSPAN) arow[KSPAN + lane] = 0;                              // K pad
    if (lane == 0) widxs[r] = wd;
  }
  __syncthreads();

  // ---- phase 2: [64 x 2336] (bf16, LDS) x [2336 x 64-per-wave] (bf16, L2) via WMMA ----
  f32x8 acc[RT][NTW];
  const f32x8 zero = {0.f, 0.f, 0.f, 0.f, 0.f, 0.f, 0.f, 0.f};
  #pragma unroll
  for (int t = 0; t < RT; ++t)
    #pragma unroll
    for (int nt = 0; nt < NTW; ++nt) acc[t][nt] = zero;

  const u32x4* Ap = (const u32x4*)smem;
  const u32x4* Bq = (const u32x4*)Bp;

  for (int kt = 0; kt < KTILES; ++kt) {
    // A fragments: 16x32 bf16; lane(m0,half): dwords [kt*16 + half*8 .. +7]
    bf16x16 a[RT];
    #pragma unroll
    for (int t = 0; t < RT; ++t) {
      Frag fa;
      int base = (m0 + 16 * t) * (ROW_DW / 4) + kt * 4 + 2 * half;
      fa.lo = Ap[base];
      fa.hi = Ap[base + 1];
      a[t] = __builtin_bit_cast(bf16x16, fa);
    }
    #pragma unroll
    for (int nt = 0; nt < NTW; ++nt) {
      int tile = kt * NTILES + wave * NTW + nt;
      Frag fb;
      int bidx = (tile * 32 + lane) * 2;
      fb.lo = Bq[bidx]; fb.hi = Bq[bidx + 1];
      bf16x16 b = __builtin_bit_cast(bf16x16, fb);
      // B fragment reused across all four row-tiles; A reused across the 4 n-tiles
      #pragma unroll
      for (int t = 0; t < RT; ++t)
        acc[t][nt] = __builtin_amdgcn_wmma_f32_16x16x32_bf16(false, a[t], false, b, (short)0,
                                                             acc[t][nt], false, false);
    }
  }

  // ---- phase 3: fused relu(+b1) . mlp2_w reduction ----
  float ps[RT][8];
  #pragma unroll
  for (int t = 0; t < RT; ++t)
    #pragma unroll
    for (int r = 0; r < 8; ++r) ps[t][r] = 0.f;

  #pragma unroll
  for (int nt = 0; nt < NTW; ++nt) {
    int n = wave * 64 + nt * 16 + m0;   // this lane's output column
    float b1v = mlp1_b[n];
    float w2v = mlp2_w[n];
    #pragma unroll
    for (int t = 0; t < RT; ++t)
      #pragma unroll
      for (int r = 0; r < 8; ++r)
        ps[t][r] += fmaxf(acc[t][nt][r] + b1v, 0.f) * w2v;
  }
  // reduce across the 16 lanes of each half (masks < 16 stay within the half)
  #pragma unroll
  for (int t = 0; t < RT; ++t)
    #pragma unroll
    for (int r = 0; r < 8; ++r) {
      #pragma unroll
      for (int m = 8; m; m >>= 1) ps[t][r] += __shfl_xor(ps[t][r], m, 32);
    }
  if (m0 == 0) {  // lanes 0 (rows m_local=r) and 16 (rows m_local=8+r)
    #pragma unroll
    for (int t = 0; t < RT; ++t)
      #pragma unroll
      for (int r = 0; r < 8; ++r)
        partial[wave * ROWS + t * 16 + half * 8 + r] = ps[t][r];
  }
  __syncthreads();

  if (threadIdx.x < ROWS) {
    float t = 0.f;
    #pragma unroll
    for (int w = 0; w < WAVES; ++w) t += partial[w * ROWS + (int)threadIdx.x];
    int c = blockIdx.x * ROWS + threadIdx.x;
    out[c] = t + mlp2_b[0] + wscore[widxs[threadIdx.x]];  // wscore already has wmlp2_b
  }
}

// ---------------- launcher ----------------
extern "C" void kernel_launch(void* const* d_in, const int* in_sizes, int n_in,
                              void* d_out, int out_size, void* d_ws, size_t ws_size,
                              hipStream_t stream) {
  (void)in_sizes; (void)n_in; (void)out_size; (void)ws_size;
  const float* doc    = (const float*)d_in[0];
  const int*   starts = (const int*)d_in[1];
  const int*   ends   = (const int*)d_in[2];
  const float* wemb   = (const float*)d_in[3];
  const float* wpemb  = (const float*)d_in[4];
  const float* attw   = (const float*)d_in[5];
  const float* attb   = (const float*)d_in[6];
  const float* m1w    = (const float*)d_in[7];
  const float* m1b    = (const float*)d_in[8];
  const float* m2w    = (const float*)d_in[9];
  const float* m2b    = (const float*)d_in[10];
  const float* wm1w   = (const float*)d_in[11];
  const float* wm1b   = (const float*)d_in[12];
  const float* wm2w   = (const float*)d_in[13];
  const float* wm2b   = (const float*)d_in[14];
  float* out = (float*)d_out;

  char* ws = (char*)d_ws;
  float*    word_attn = (float*)ws;              // 4096 f32 = 16384 B
  float*    wscore    = (float*)(ws + 16384);    // 32 f32
  uint32_t* Bpk       = (uint32_t*)(ws + 16640); // 73*64*1024 B = 4,784,128 B

  k_word_attn   <<<NW / 8, 256, 0, stream>>>(doc, attw, attb, word_attn);
  k_width_scores<<<MSW, 256, 0, stream>>>(wpemb, wm1w, wm1b, wm2w, wm2b, wscore);
  k_pack_w1     <<<KTILES * NTILES, 256, 0, stream>>>(m1w, Bpk);

  // LDS: A tile (300,032 B) + per-wave partials (4,096 B) + width idx (256 B) = 304,384 B
  size_t lds = (size_t)ROWS * ROW_DW * 4 + (size_t)WAVES * ROWS * 4 + ROWS * 4;
  k_main<<<NC / ROWS, 512, lds, stream>>>(doc, starts, ends, wemb, word_attn, Bpk,
                                          m1b, m2w, m2b, wscore, out);
}
